// Net_45887430590898
// MI455X (gfx1250) — matrix operand order repaired and checked by reading, not compiled, not run on hardware
//
#include <hip/hip_runtime.h>
#include <math.h>

// ---------------- problem constants ----------------
#define NN 3072      // nodes
#define NE 30720     // edges
#define NG 16        // graphs
#define NT 6         // towers
#define FLT_BIG 3.402823466e38f

typedef __attribute__((ext_vector_type(16))) _Float16 v16h;
typedef __attribute__((ext_vector_type(8)))  _Float16 v8h;
typedef __attribute__((ext_vector_type(8)))  float    v8f;

#define SHUF16(a0, a1) __builtin_shufflevector(a0, a1, 0, 1, 2, 3, 4, 5, 6, 7, 8, 9, 10, 11, 12, 13, 14, 15)

static inline int cdiv(int a, int b) { return (a + b - 1) / b; }

// ---------------- atomics helpers ----------------
__device__ inline void atomic_maxf(float* addr, float val) {
  unsigned int* u = (unsigned int*)addr;
  unsigned int old = *u;
  while (__uint_as_float(old) < val) {
    unsigned int assumed = old;
    old = atomicCAS(u, assumed, __float_as_uint(val));
    if (old == assumed) break;
  }
}
__device__ inline void atomic_minf(float* addr, float val) {
  unsigned int* u = (unsigned int*)addr;
  unsigned int old = *u;
  while (__uint_as_float(old) > val) {
    unsigned int assumed = old;
    old = atomicCAS(u, assumed, __float_as_uint(val));
    if (old == assumed) break;
  }
}

// ---------------- small elementwise kernels ----------------
__global__ void k_fill(float* p, int n, float v) {
  int i = blockIdx.x * blockDim.x + threadIdx.x;
  if (i < n) p[i] = v;
}
// f32 [rows,srcW] -> f16 [rows,dstW] zero padded
__global__ void k_pad_f16(const float* __restrict__ s, _Float16* __restrict__ d,
                          int srcW, int dstW, int total) {
  int i = blockIdx.x * blockDim.x + threadIdx.x;
  if (i >= total) return;
  int n = i / dstW, c = i - n * dstW;
  d[i] = (_Float16)((c < srcW) ? s[n * srcW + c] : 0.0f);
}
__global__ void k_deg(const int* __restrict__ dst, float* deg) {
  int i = blockIdx.x * blockDim.x + threadIdx.x;
  if (i < NE) atomicAdd(deg + dst[i], 1.0f);
}
__global__ void k_avglog(const float* __restrict__ hist, float* out) {
  __shared__ float sn[64], sd[64];
  int i = threadIdx.x;
  float w = hist[i];
  sn[i] = logf((float)(i + 1)) * w;
  sd[i] = w;
  __syncthreads();
  for (int s = 32; s > 0; s >>= 1) {
    if (i < s) { sn[i] += sn[i + s]; sd[i] += sd[i + s]; }
    __syncthreads();
  }
  if (i == 0) out[0] = sn[0] / sd[0];
}
__global__ void k_scalers(float* degc, float* amp, float* att, const float* avgl) {
  int i = blockIdx.x * blockDim.x + threadIdx.x;
  if (i >= NN) return;
  float d = fmaxf(degc[i], 1.0f);
  degc[i] = d;
  float l = logf(d + 1.0f);
  amp[i] = l / avgl[0];
  att[i] = avgl[0] / l;
}
// e = edge_attr @ ew + eb -> f16 [E, segW] zero padded
__global__ void k_edge_proj(const float* __restrict__ ea, const float* __restrict__ ew,
                            const float* __restrict__ eb, _Float16* __restrict__ ef,
                            int F, int segW) {
  int i = blockIdx.x * blockDim.x + threadIdx.x;
  if (i >= NE * segW) return;
  int e = i / segW, f = i - e * segW;
  float v = 0.0f;
  if (f < F) v = ea[e * 2] * ew[f] + ea[e * 2 + 1] * ew[F + f] + eb[f];
  ef[i] = (_Float16)v;
}
// prew [T,3F,F] -> swizzled B panel, fragment order [kt][nt][lane][16]
__global__ void k_sw_prew(const float* __restrict__ w, _Float16* __restrict__ B,
                          int F, int Fp, int segW, int KT, int NTt) {
  int id = blockIdx.x * blockDim.x + threadIdx.x;
  if (id >= KT * NTt * 512) return;
  int tile = id >> 9, r = id & 511;
  int lane = r >> 4, i = r & 15;
  int kt = tile / NTt, nt = tile - kt * NTt;
  int k = kt * 32 + ((lane >> 4) << 3) + (i & 7) + ((i >> 3) << 4);
  int n = nt * 16 + (lane & 15);
  int seg = k / segW, idx = k - seg * segW;
  int t = n / Fp, f = n - t * Fp;
  float v = 0.0f;
  if (seg < 3 && idx < F && f < F) v = w[((t * 3 * F) + seg * F + idx) * F + f];
  B[id] = (_Float16)v;
}
__global__ void k_pre_bias(const float* __restrict__ b, float* __restrict__ o,
                           int F, int Fp, int Np) {
  int i = blockIdx.x * blockDim.x + threadIdx.x;
  if (i >= Np) return;
  int t = i / Fp, f = i - t * Fp;
  o[i] = (f < F) ? b[t * F + f] : 0.0f;
}
// postw [T,13F,50] -> swizzled B panel; K: 13 segments x segW; N: T towers x 64 (f<50)
__global__ void k_sw_postw(const float* __restrict__ w, _Float16* __restrict__ B,
                           int F, int segW, int KT, int NTt) {
  int id = blockIdx.x * blockDim.x + threadIdx.x;
  if (id >= KT * NTt * 512) return;
  int tile = id >> 9, r = id & 511;
  int lane = r >> 4, i = r & 15;
  int kt = tile / NTt, nt = tile - kt * NTt;
  int k = kt * 32 + ((lane >> 4) << 3) + (i & 7) + ((i >> 3) << 4);
  int n = nt * 16 + (lane & 15);
  int seg = k / segW, idx = k - seg * segW;
  int t = n >> 6, f = n & 63;
  int c = (seg == 0) ? idx : (F + (seg - 1) * F + idx);
  float v = 0.0f;
  if (idx < F && f < 50) v = w[(t * 13 * F + c) * 50 + f];
  B[id] = (_Float16)v;
}
// linw [300,300] -> swizzled [KT=10][NTt=19] fragment panel
__global__ void k_sw_linw(const float* __restrict__ w, _Float16* __restrict__ B) {
  int id = blockIdx.x * blockDim.x + threadIdx.x;
  if (id >= 10 * 19 * 512) return;
  int tile = id >> 9, r = id & 511;
  int lane = r >> 4, i = r & 15;
  int kt = tile / 19, nt = tile - kt * 19;
  int k = kt * 32 + ((lane >> 4) << 3) + (i & 7) + ((i >> 3) << 4);
  int n = nt * 16 + (lane & 15);
  B[id] = (_Float16)((k < 300 && n < 300) ? w[k * 300 + n] : 0.0f);
}
// postout f32 [N,304] -> lin input f16 [N,320]
__global__ void k_linin(const float* __restrict__ p, _Float16* __restrict__ d) {
  int i = blockIdx.x * blockDim.x + threadIdx.x;
  if (i >= NN * 320) return;
  int n = i / 320, c = i - n * 320;
  d[i] = (_Float16)((c < 300) ? p[n * 304 + c] : 0.0f);
}
// s1/s2/min/max [N, T*Fp] -> aggf16 [N, T, 12, segW]: (mean,mn,mx,std) x {1, amp, att}
__global__ void k_agg(const float* __restrict__ s1, const float* __restrict__ s2,
                      const float* __restrict__ mnb, const float* __restrict__ mxb,
                      const float* __restrict__ degc, const float* __restrict__ amp,
                      const float* __restrict__ att, _Float16* __restrict__ agg,
                      int F, int Fp, int segW) {
  int i = blockIdx.x * blockDim.x + threadIdx.x;
  if (i >= NN * NT * segW) return;
  int n = i / (NT * segW), rem = i - n * NT * segW;
  int t = rem / segW, f = rem - t * segW;
  float st0 = 0.f, st1 = 0.f, st2 = 0.f, st3 = 0.f;
  if (f < F) {
    int base = n * (NT * Fp) + t * Fp + f;
    float d = degc[n];
    float m1 = s1[base] / d;
    float var = fmaxf(s2[base] / d - m1 * m1, 0.0f);
    st0 = m1; st1 = mnb[base]; st2 = mxb[base]; st3 = sqrtf(var + 1e-5f);
  }
  float stats[4] = {st0, st1, st2, st3};
  float scl[3] = {1.0f, amp[n], att[n]};
  size_t ob = ((size_t)(n * NT + t) * 12) * segW + f;
#pragma unroll
  for (int g = 0; g < 3; ++g)
#pragma unroll
    for (int s = 0; s < 4; ++s)
      agg[ob + (size_t)(g * 4 + s) * segW] = (_Float16)(scl[g] * stats[s]);
}

// ---------------- GEMM 1: edge pre-MLP, fused segment aggregation ----------------
// Compile-time shapes: no remainder paths, constant strides/immediates.
template <int SEGW, int TPS, int NP, int NTILES>
__global__ __launch_bounds__(256) void pre_gemm(
    const _Float16* __restrict__ hf, const _Float16* __restrict__ ef,
    const v16h* __restrict__ Bsw, const float* __restrict__ bias,
    const int* __restrict__ srcI, const int* __restrict__ dstI,
    float* __restrict__ s1, float* __restrict__ s2,
    float* __restrict__ mnb, float* __restrict__ mxb) {
  int tile = blockIdx.x * blockDim.y + threadIdx.y;
  if (tile >= (NE / 16) * NTILES) return;     // whole-wave exit: EXEC all-1s for WMMA
  int mt = tile / NTILES, nt = tile - mt * NTILES;
  int lane = threadIdx.x, half = lane >> 4, m = lane & 15;
  int row0 = mt * 16, col0 = nt * 16;
  int e = row0 + m;
  const _Float16* bases[3];
  bases[0] = hf + (size_t)dstI[e] * SEGW;
  bases[1] = hf + (size_t)srcI[e] * SEGW;
  bases[2] = ef + (size_t)e * SEGW;
  int hoff = half * 8;
  const v16h* bp = Bsw + (size_t)nt * 32 + lane;
  v8f acc = {};
#pragma unroll
  for (int seg = 0; seg < 3; ++seg) {
    const _Float16* p = bases[seg] + hoff;
#pragma unroll 2
    for (int j = 0; j < TPS; ++j) {
      v8h a0 = *(const v8h*)(p + j * 32);
      v8h a1 = *(const v8h*)(p + j * 32 + 16);
      v16h a = SHUF16(a0, a1);
      __builtin_prefetch((const void*)(bp + (size_t)NTILES * 32), 0, 3);
      v16h b = *bp;
      bp += (size_t)NTILES * 32;
      acc = __builtin_amdgcn_wmma_f32_16x16x32_f16(false, a, false, b, (short)0, acc,
                                                   false, false);
    }
  }
  int n = col0 + m;
  float bv = bias[n];
  int mh = half * 8;
#pragma unroll
  for (int r = 0; r < 8; ++r) {
    int erow = row0 + mh + r;
    float v = acc[r] + bv;
    long base = (long)dstI[erow] * NP + n;
    atomicAdd(s1 + base, v);
    atomicAdd(s2 + base, v * v);
    atomic_maxf(mxb + base, v);
    atomic_minf(mnb + base, v);
  }
}

// ---------------- GEMM 2: post-MLP (13 segments -> 50 per tower) ----------------
template <int SEGW, int TPS>
__global__ __launch_bounds__(256) void post_gemm(
    const _Float16* __restrict__ hf, const _Float16* __restrict__ agg,
    const v16h* __restrict__ Bsw, const float* __restrict__ postb,
    float* __restrict__ out) {
  constexpr int TPR = NT * 4;                  // 24 col tiles (t*64 + ct*16)
  int tile = blockIdx.x * blockDim.y + threadIdx.y;
  if (tile >= (NN / 16) * TPR) return;
  int mt = tile / TPR, rem = tile - mt * TPR;
  int t = rem >> 2, ct = rem & 3;
  int lane = threadIdx.x, half = lane >> 4, m = lane & 15;
  int row0 = mt * 16;
  int nrow = row0 + m;
  int hoff = half * 8;
  const _Float16* hp = hf + (size_t)nrow * SEGW + hoff;
  const _Float16* ap = agg + ((size_t)(nrow * NT + t) * 12) * SEGW + hoff;
  const v16h* bp = Bsw + (size_t)rem * 32 + lane;   // rem == B col-tile index
  v8f acc = {};
  // segment 0: h
#pragma unroll 2
  for (int j = 0; j < TPS; ++j) {
    v8h a0 = *(const v8h*)(hp + j * 32);
    v8h a1 = *(const v8h*)(hp + j * 32 + 16);
    v16h a = SHUF16(a0, a1);
    v16h b = *bp;
    bp += (size_t)TPR * 32;
    acc = __builtin_amdgcn_wmma_f32_16x16x32_f16(false, a, false, b, (short)0, acc,
                                                 false, false);
  }
  // segments 1..12: pre-scaled aggregates
#pragma unroll
  for (int sub = 0; sub < 12; ++sub) {
    const _Float16* p = ap + (size_t)sub * SEGW;
#pragma unroll 2
    for (int j = 0; j < TPS; ++j) {
      v8h a0 = *(const v8h*)(p + j * 32);
      v8h a1 = *(const v8h*)(p + j * 32 + 16);
      v16h a = SHUF16(a0, a1);
      __builtin_prefetch((const void*)(bp + (size_t)TPR * 32), 0, 3);
      v16h b = *bp;
      bp += (size_t)TPR * 32;
      acc = __builtin_amdgcn_wmma_f32_16x16x32_f16(false, a, false, b, (short)0, acc,
                                                   false, false);
    }
  }
  int f = ct * 16 + m;
  if (f < 50) {
    float bv = postb[t * 50 + f];
    int mh = half * 8;
#pragma unroll
    for (int r = 0; r < 8; ++r)
      out[(long)(row0 + mh + r) * 304 + t * 50 + f] = acc[r] + bv;
  }
}

// ---------------- GEMM 3: node linear 300x300 (+bias, optional ReLU) ------------
__global__ __launch_bounds__(256) void lin_gemm(
    const _Float16* __restrict__ A, const v16h* __restrict__ Bsw,
    const float* __restrict__ bias, float* __restrict__ out, int relu) {
  int tile = blockIdx.x * blockDim.y + threadIdx.y;
  if (tile >= (NN / 16) * 19) return;
  int mt = tile / 19, nt = tile - mt * 19;
  int lane = threadIdx.x, half = lane >> 4, m = lane & 15;
  int row0 = mt * 16, col0 = nt * 16;
  int hoff = half * 8;
  const _Float16* ap = A + (size_t)(row0 + m) * 320 + hoff;
  const v16h* bp = Bsw + (size_t)nt * 32 + lane;
  v8f acc = {};
#pragma unroll 2
  for (int kt = 0; kt < 10; ++kt) {
    v8h a0 = *(const v8h*)(ap + kt * 32);
    v8h a1 = *(const v8h*)(ap + kt * 32 + 16);
    v16h a = SHUF16(a0, a1);
    v16h b = *bp;
    bp += (size_t)19 * 32;
    acc = __builtin_amdgcn_wmma_f32_16x16x32_f16(false, a, false, b, (short)0, acc,
                                                 false, false);
  }
  int col = col0 + m;
  float bv = (col < 300) ? bias[col] : 0.0f;
  int mh = half * 8;
#pragma unroll
  for (int r = 0; r < 8; ++r) {
    float v = acc[r] + bv;
    if (relu) v = fmaxf(v, 0.0f);
    out[(long)(row0 + mh + r) * 304 + col] = v;
  }
}

// ---------------- pooling + prediction head ----------------
__global__ void k_pool(const float* __restrict__ h, float* __restrict__ g) {
  int gi = blockIdx.x, c = threadIdx.x;   // 16 blocks x 304 threads
  float s = 0.0f;
  for (int r = 0; r < NN / NG; ++r) s += h[((long)gi * (NN / NG) + r) * 304 + c];
  g[gi * 304 + c] = s;
}
__global__ void k_head(const float* __restrict__ in, const float* __restrict__ w,
                       const float* __restrict__ b, float* __restrict__ out,
                       int inDim, int inStride, int outDim) {
  int id = blockIdx.x * blockDim.x + threadIdx.x;
  if (id >= NG * outDim) return;
  int gi = id / outDim, j = id - gi * outDim;
  float s = b[j];
  for (int k = 0; k < inDim; ++k) s += in[gi * inStride + k] * w[k * outDim + j];
  out[gi * outDim + j] = fmaxf(s, 0.0f);
}

// ---------------- host orchestration ----------------
extern "C" void kernel_launch(void* const* d_in, const int* in_sizes, int n_in,
                              void* d_out, int out_size, void* d_ws, size_t ws_size,
                              hipStream_t stream) {
  (void)in_sizes; (void)n_in; (void)out_size; (void)ws_size;
  const float* x  = (const float*)d_in[0];
  const float* ea = (const float*)d_in[1];
  const int*   ei = (const int*)d_in[2];
  const int* srcI = ei;
  const int* dstI = ei + NE;
  const float* dh = (const float*)d_in[4];
  const float* hw1 = (const float*)d_in[21]; const float* hb1 = (const float*)d_in[22];
  const float* hw2 = (const float*)d_in[23]; const float* hb2 = (const float*)d_in[24];
  const float* hw3 = (const float*)d_in[25]; const float* hb3 = (const float*)d_in[26];

  char* ws = (char*)d_ws;
  size_t off = 0;
  auto alloc = [&](size_t bytes) -> char* {
    char* p = ws + off;
    off += (bytes + 255) & ~(size_t)255;
    return p;
  };
  float*     degc = (float*)alloc(NN * 4);
  float*     ampv = (float*)alloc(NN * 4);
  float*     attv = (float*)alloc(NN * 4);
  float*     avgl = (float*)alloc(256);
  _Float16*  hf   = (_Float16*)alloc((size_t)NN * 320 * 2);
  _Float16*  ef   = (_Float16*)alloc((size_t)NE * 320 * 2);
  float*     hbuf = (float*)alloc((size_t)NN * 304 * 4);
  float*     pout = (float*)alloc((size_t)NN * 304 * 4);
  _Float16*  lini = (_Float16*)alloc((size_t)NN * 320 * 2);
  float*     s1   = (float*)alloc((size_t)NN * 1824 * 4);
  float*     s2   = (float*)alloc((size_t)NN * 1824 * 4);
  float*     mnb  = (float*)alloc((size_t)NN * 1824 * 4);
  float*     mxb  = (float*)alloc((size_t)NN * 1824 * 4);
  _Float16*  aggf = (_Float16*)alloc((size_t)NN * NT * 12 * 320 * 2);
  _Float16*  Bpre = (_Float16*)alloc((size_t)30 * 114 * 512 * 2);
  float*     bpre = (float*)alloc(1824 * 4);
  _Float16*  Bpost= (_Float16*)alloc((size_t)130 * 24 * 512 * 2);
  _Float16*  Blin = (_Float16*)alloc((size_t)10 * 19 * 512 * 2);
  float*     g    = (float*)alloc(NG * 304 * 4);
  float*     o1   = (float*)alloc(NG * 600 * 4);
  float*     o2   = (float*)alloc(NG * 304 * 4);

  const int TB = 256;
  // degree scalers
  k_fill<<<cdiv(NN, TB), TB, 0, stream>>>(degc, NN, 0.0f);
  k_deg<<<cdiv(NE, TB), TB, 0, stream>>>(dstI, degc);
  k_avglog<<<1, 64, 0, stream>>>(dh, avgl);
  k_scalers<<<cdiv(NN, TB), TB, 0, stream>>>(degc, ampv, attv, avgl);

  dim3 blk(32, 8);
  for (int l = 0; l < 5; ++l) {
    int F, Fp, segW, tps, Np, ntPre, relu;
    const float *ew, *ebp, *prew, *preb, *postw, *postb, *linw, *linb;
    if (l == 0) {
      F = 32; Fp = 32; segW = 32; tps = 1; Np = 192; ntPre = 12; relu = 1;
      ew   = (const float*)d_in[5];  ebp  = (const float*)d_in[6];
      prew = (const float*)d_in[7];  preb = (const float*)d_in[8];
      postw= (const float*)d_in[9];  postb= (const float*)d_in[10];
      linw = (const float*)d_in[11]; linb = (const float*)d_in[12];
    } else {
      int li = l - 1;
      F = 300; Fp = 304; segW = 320; tps = 10; Np = 1824; ntPre = 114;
      relu = (li < 3) ? 1 : 0;
      ew   = (const float*)d_in[13] + (size_t)li * 2 * 300;
      ebp  = (const float*)d_in[14] + (size_t)li * 300;
      prew = (const float*)d_in[15] + (size_t)li * NT * 900 * 300;
      preb = (const float*)d_in[16] + (size_t)li * NT * 300;
      postw= (const float*)d_in[17] + (size_t)li * NT * 3900 * 50;
      postb= (const float*)d_in[18] + (size_t)li * NT * 50;
      linw = (const float*)d_in[19] + (size_t)li * 300 * 300;
      linb = (const float*)d_in[20] + (size_t)li * 300;
    }
    int KTpre = 3 * tps, KTpost = 13 * tps;

    // activations + weights -> f16 (zero padded / fragment-swizzled)
    if (l == 0)
      k_pad_f16<<<cdiv(NN * segW, TB), TB, 0, stream>>>(x, hf, 32, segW, NN * segW);
    else
      k_pad_f16<<<cdiv(NN * segW, TB), TB, 0, stream>>>(hbuf, hf, 304, segW, NN * segW);
    k_edge_proj<<<cdiv(NE * segW, TB), TB, 0, stream>>>(ea, ew, ebp, ef, F, segW);
    k_sw_prew<<<cdiv(KTpre * ntPre * 512, TB), TB, 0, stream>>>(prew, Bpre, F, Fp, segW,
                                                                KTpre, ntPre);
    k_pre_bias<<<cdiv(Np, TB), TB, 0, stream>>>(preb, bpre, F, Fp, Np);

    // init aggregation buffers
    int segN = NN * Np;
    k_fill<<<cdiv(segN, TB), TB, 0, stream>>>(s1, segN, 0.0f);
    k_fill<<<cdiv(segN, TB), TB, 0, stream>>>(s2, segN, 0.0f);
    k_fill<<<cdiv(segN, TB), TB, 0, stream>>>(mnb, segN, FLT_BIG);
    k_fill<<<cdiv(segN, TB), TB, 0, stream>>>(mxb, segN, -FLT_BIG);

    // edge GEMM + fused segment sum/sumsq/min/max
    if (l == 0) {
      int tilesPre = (NE / 16) * 12;
      pre_gemm<32, 1, 192, 12><<<cdiv(tilesPre, 8), blk, 0, stream>>>(
          hf, ef, (const v16h*)Bpre, bpre, srcI, dstI, s1, s2, mnb, mxb);
    } else {
      int tilesPre = (NE / 16) * 114;
      pre_gemm<320, 10, 1824, 114><<<cdiv(tilesPre, 8), blk, 0, stream>>>(
          hf, ef, (const v16h*)Bpre, bpre, srcI, dstI, s1, s2, mnb, mxb);
    }
    // per-node aggregates -> pre-scaled f16 blocks
    k_agg<<<cdiv(NN * NT * segW, TB), TB, 0, stream>>>(s1, s2, mnb, mxb, degc, ampv, attv,
                                                       aggf, F, Fp, segW);
    // post-MLP GEMM
    k_sw_postw<<<cdiv(KTpost * 24 * 512, TB), TB, 0, stream>>>(postw, Bpost, F, segW,
                                                               KTpost, 24);
    int tilesPost = (NN / 16) * 24;
    if (l == 0)
      post_gemm<32, 1><<<cdiv(tilesPost, 8), blk, 0, stream>>>(
          hf, aggf, (const v16h*)Bpost, postb, pout);
    else
      post_gemm<320, 10><<<cdiv(tilesPost, 8), blk, 0, stream>>>(
          hf, aggf, (const v16h*)Bpost, postb, pout);
    // lin GEMM
    k_linin<<<cdiv(NN * 320, TB), TB, 0, stream>>>(pout, lini);
    k_sw_linw<<<cdiv(10 * 19 * 512, TB), TB, 0, stream>>>(linw, Blin);
    int tilesLin = (NN / 16) * 19;
    lin_gemm<<<cdiv(tilesLin, 8), blk, 0, stream>>>(lini, (const v16h*)Blin, linb,
                                                    hbuf, relu);
  }

  // global add pool + head
  k_pool<<<NG, 304, 0, stream>>>(hbuf, g);
  k_head<<<cdiv(NG * 600, 128), 128, 0, stream>>>(g, hw1, hb1, o1, 300, 304, 600);
  k_head<<<cdiv(NG * 300, 128), 128, 0, stream>>>(o1, hw2, hb2, o2, 600, 600, 300);
  k_head<<<cdiv(NG, 64), 64, 0, stream>>>(o2, hw3, hb3, (float*)d_out, 300, 300, 1);
}
